// DipolePredictorSE3_35948876267612
// MI455X (gfx1250) — compile-verified, matrix operand order checked by reading, not run on hardware
//
#include <hip/hip_runtime.h>
#include <hip/hip_bf16.h>

// Problem constants (match reference)
#define Bq 4
#define Nq 2048
#define F_INq 3
#define Hq 32
#define H2q 16
#define Lq 3
#define BN (Bq * Nq)

typedef __attribute__((ext_vector_type(16))) _Float16 v16h;
typedef __attribute__((ext_vector_type(8)))  _Float16 v8h;
typedef __attribute__((ext_vector_type(8)))  float    v8f;
typedef __attribute__((ext_vector_type(4)))  float    v4f;
typedef __attribute__((ext_vector_type(2)))  float    v2f;

#define NEGMAX (-3.402823466e38f)

// Load a 16-half WMMA operand (A layout / assumed matching B layout):
// element t -> K = 16*(t>>3) + 8*khalf + (t&7), i.e. two contiguous 8-half
// (16-byte) chunks at half-offsets 8*kh and 16+8*kh from the row base.
__device__ __forceinline__ v16h load_tile_row(const _Float16* __restrict__ rowp, int kh) {
  const v8h c0 = *(const v8h*)(rowp + 8 * kh);
  const v8h c1 = *(const v8h*)(rowp + 16 + 8 * kh);
  v16h r;
#pragma unroll
  for (int t = 0; t < 8; ++t) { r[t] = c0[t]; r[t + 8] = c1[t]; }
  return r;
}

// ---------------------------------------------------------------------------
// h = feats @ W_in ; x0 = coors
__global__ __launch_bounds__(256) void k_embed(
    const float* __restrict__ feats, const float* __restrict__ coors,
    const float* __restrict__ W_in, float* __restrict__ h, float* __restrict__ x0) {
  int idx = blockIdx.x * blockDim.x + threadIdx.x;   // over BN*H
  int col = idx & (Hq - 1);
  long row = idx >> 5;
  float s = 0.f;
#pragma unroll
  for (int e = 0; e < F_INq; ++e) s += feats[row * F_INq + e] * W_in[e * Hq + col];
  h[idx] = s;
  if (col < 3) x0[row * 3 + col] = coors[row * 3 + col];
}

// ---------------------------------------------------------------------------
// q,k in f16 row-major [BN,H] (q pre-scaled by 1/sqrt(H)); v transposed [B,H,N] f16
__global__ __launch_bounds__(256) void k_qkv(
    const float* __restrict__ h, const float* __restrict__ Wq,
    const float* __restrict__ Wk, const float* __restrict__ Wv,
    _Float16* __restrict__ qf, _Float16* __restrict__ kf, _Float16* __restrict__ vT) {
  int idx = blockIdx.x * blockDim.x + threadIdx.x;   // over BN*H
  int col = idx & (Hq - 1);
  long row = idx >> 5;
  const float* hr = h + row * Hq;
  float sq = 0.f, sk = 0.f, sv = 0.f;
#pragma unroll
  for (int e = 0; e < Hq; ++e) {
    float he = hr[e];
    sq += he * Wq[e * Hq + col];
    sk += he * Wk[e * Hq + col];
    sv += he * Wv[e * Hq + col];
  }
  qf[idx] = (_Float16)(sq * 0.17677669529663687f);   // 1/sqrt(32)
  kf[idx] = (_Float16)sk;
  int b = (int)(row >> 11);                          // N = 2048
  int n = (int)(row & (Nq - 1));
  vT[((long)b * Hq + col) * Nq + n] = (_Float16)sv;
}

// ---------------------------------------------------------------------------
// Per-layer geometry packs from current coordinates x:
//   g4 [BN,4] f32 = {x0,x1,x2, x.x}            (A-side of the K=4 f32 WMMA)
//   gq4[BN,4] f32 = {2x0,2x1,2x2, -1}          (B-side)
//   xT [B,16,N] f16 = coords transposed, rows 3..15 zero (A-side of coord PV)
__global__ __launch_bounds__(256) void k_geo(
    const float* __restrict__ x_in, float* __restrict__ g4,
    float* __restrict__ gq4, _Float16* __restrict__ xT) {
  int idx = blockIdx.x * blockDim.x + threadIdx.x;   // over B*16*N
  int n = idx & (Nq - 1);
  int comp = (idx >> 11) & 15;
  int b = idx >> 15;
  const float* xr = x_in + ((long)b * Nq + n) * 3;
  float val = (comp < 3) ? xr[comp] : 0.f;
  xT[idx] = (_Float16)val;
  if (comp == 0) {
    float x0 = xr[0], x1 = xr[1], x2 = xr[2];
    long r = (long)b * Nq + n;
    float* g = g4 + r * 4;
    g[0] = x0; g[1] = x1; g[2] = x2; g[3] = x0 * x0 + x1 * x1 + x2 * x2;
    float* gq = gq4 + r * 4;
    gq[0] = 2.f * x0; gq[1] = 2.f * x1; gq[2] = 2.f * x2; gq[3] = -1.f;
  }
}

// ---------------------------------------------------------------------------
// Fused flash attention + equivariant coordinate update, all matmuls on WMMA:
//   Sᵀ  = K_tile × Qᵀ            (f16 WMMA, K=32 == H)
//   Sᵀ += G_tile × Gqᵀ           (f32 WMMA, K=4: 2xi.xj - |xj|²; -|xi|² dropped
//                                 since softmax is shift-invariant per row)
//   Oᵀ += Vᵀ × Pᵀ ; Cᵀ += Xᵀ × Pᵀ (f16 WMMAs; Pᵀ D-tiles feed B operand directly)
__global__ __launch_bounds__(32) void k_attn(
    const _Float16* __restrict__ qf, const _Float16* __restrict__ kf,
    const _Float16* __restrict__ vT, const float* __restrict__ g4,
    const float* __restrict__ gq4, const _Float16* __restrict__ xT,
    const float* __restrict__ x_in, const unsigned char* __restrict__ adj,
    const float* __restrict__ coor_scale_l,
    float* __restrict__ ao, float* __restrict__ x_out) {
  const int lane = threadIdx.x;
  const int m = lane & 15;        // query column i within tile / operand row
  const int kh = lane >> 4;       // half-wave: selects K-chunk / j-subrange
  const int tiles = Nq / 16;
  const int b = blockIdx.x / tiles;
  const int itile = (blockIdx.x % tiles) * 16;
  const long rowg = (long)b * Nq + itile + m;

  // per-i operands (loaded once)
  const v16h qB = load_tile_row(qf + rowg * Hq, kh);
  const v2f  gB = *(const v2f*)(gq4 + rowg * 4 + 2 * kh);

  const float xi0 = x_in[rowg * 3 + 0];
  const float xi1 = x_in[rowg * 3 + 1];
  const float xi2 = x_in[rowg * 3 + 2];
  const unsigned char* arow = adj + rowg * (long)Nq;

  v8f Oa = {};            // O^T rows h = 0..15
  v8f Ob = {};            // O^T rows h = 16..31
  v8f Oc = {};            // C^T rows comp = 0..15 (3 real + zero pad)
  float mrun = NEGMAX, lrun = 0.f;

  const _Float16* kbase = kf + (long)b * Nq * Hq;
  const float*    gbase = g4 + (long)b * Nq * 4;
  const _Float16* vrow0 = vT + ((long)b * Hq + m) * Nq;
  const _Float16* vrow1 = vT + ((long)b * Hq + 16 + m) * Nq;
  const _Float16* xrow  = xT + ((long)b * 16 + m) * Nq;

  for (int jb = 0; jb < Nq; jb += 32) {
    if (jb + 32 < Nq) {  // hint the next j-block toward the caches
      __builtin_prefetch(kbase + (long)(jb + 32 + m) * Hq, 0, 1);
      __builtin_prefetch(vrow0 + jb + 32, 0, 1);
    }
    // A operands: two 16-row K tiles (rows j) + geometry packs
    const v16h A0 = load_tile_row(kbase + (long)(jb + m) * Hq, kh);
    const v16h A1 = load_tile_row(kbase + (long)(jb + 16 + m) * Hq, kh);
    const v2f  G0 = *(const v2f*)(gbase + (long)(jb + m) * 4 + 2 * kh);
    const v2f  G1 = *(const v2f*)(gbase + (long)(jb + 16 + m) * 4 + 2 * kh);
    const v8f z = {};
    // Sᵀ[j][i] = Σ_e K[j][e] Q[i][e] + (2 xi.xj - |xj|²)
    v8f S0 = __builtin_amdgcn_wmma_f32_16x16x32_f16(false, A0, false, qB, (short)0, z, false, false);
    S0 = __builtin_amdgcn_wmma_f32_16x16x4_f32(false, G0, false, gB, (short)0, S0, false, false);
    v8f S1 = __builtin_amdgcn_wmma_f32_16x16x32_f16(false, A1, false, qB, (short)0, z, false, false);
    S1 = __builtin_amdgcn_wmma_f32_16x16x4_f32(false, G1, false, gB, (short)0, S1, false, false);

    // adjacency: 8 consecutive j per chunk, loaded as two u32
    const unsigned m0lo = *(const unsigned*)(arow + jb + 8 * kh);
    const unsigned m0hi = *(const unsigned*)(arow + jb + 8 * kh + 4);
    const unsigned m1lo = *(const unsigned*)(arow + jb + 16 + 8 * kh);
    const unsigned m1hi = *(const unsigned*)(arow + jb + 16 + 8 * kh + 4);

    float lg[16];
    float lmax = NEGMAX;
#pragma unroll
    for (int v = 0; v < 8; ++v) {
      const unsigned mk0 = ((v < 4 ? m0lo : m0hi) >> (8 * (v & 3))) & 0xFFu;
      const unsigned mk1 = ((v < 4 ? m1lo : m1hi) >> (8 * (v & 3))) & 0xFFu;
      float l0 = mk0 ? S0[v] : NEGMAX;
      float l1 = mk1 ? S1[v] : NEGMAX;
      lg[v] = l0; lg[8 + v] = l1;
      lmax = fmaxf(lmax, fmaxf(l0, l1));
    }
    // merge lane-pair sharing row i, then online-softmax update
    const float bmax = fmaxf(lmax, __shfl_xor(lmax, 16, 32));
    const float mnew = fmaxf(mrun, bmax);
    const float alpha = __expf(mrun - mnew);

    v16h PB;                     // Pᵀ as B operand, element t <-> K=j mapping
    float psum = 0.f;
#pragma unroll
    for (int t = 0; t < 16; ++t) {
      float p = __expf(lg[t] - mnew);
      psum += p;
      PB[t] = (_Float16)p;
    }
    lrun = lrun * alpha + psum;
#pragma unroll
    for (int v = 0; v < 8; ++v) { Oa[v] *= alpha; Ob[v] *= alpha; Oc[v] *= alpha; }

    // Oᵀ += Vᵀ × Pᵀ (two h-halves) ; Cᵀ += Xᵀ × Pᵀ
    const v16h Va = load_tile_row(vrow0 + jb, kh);
    const v16h Vb = load_tile_row(vrow1 + jb, kh);
    const v16h Xa = load_tile_row(xrow + jb, kh);
    Oa = __builtin_amdgcn_wmma_f32_16x16x32_f16(false, Va, false, PB, (short)0, Oa, false, false);
    Ob = __builtin_amdgcn_wmma_f32_16x16x32_f16(false, Vb, false, PB, (short)0, Ob, false, false);
    Oc = __builtin_amdgcn_wmma_f32_16x16x32_f16(false, Xa, false, PB, (short)0, Oc, false, false);
    mrun = mnew;
  }

  const float ltot = lrun + __shfl_xor(lrun, 16, 32);
  const float inv = 1.0f / ltot;

  // attention output (pre-Wo), row-major [BN,H]; lane owns 8+8 contiguous cols
  float* aorow = ao + rowg * Hq;
  v4f oa0 = {Oa[0] * inv, Oa[1] * inv, Oa[2] * inv, Oa[3] * inv};
  v4f oa1 = {Oa[4] * inv, Oa[5] * inv, Oa[6] * inv, Oa[7] * inv};
  v4f ob0 = {Ob[0] * inv, Ob[1] * inv, Ob[2] * inv, Ob[3] * inv};
  v4f ob1 = {Ob[4] * inv, Ob[5] * inv, Ob[6] * inv, Ob[7] * inv};
  *(v4f*)(aorow + 8 * kh)          = oa0;
  *(v4f*)(aorow + 8 * kh + 4)      = oa1;
  *(v4f*)(aorow + 16 + 8 * kh)     = ob0;
  *(v4f*)(aorow + 16 + 8 * kh + 4) = ob1;

  // x' = x + s * (x - attn@x)  (softmax rows sum to 1); WMMA already reduced
  // over the full K blocks, so Oc needs no cross-lane merge.
  const float sc = *coor_scale_l;
  if (kh == 0) {
    float* xo = x_out + rowg * 3;
    xo[0] = xi0 + sc * (xi0 - Oc[0] * inv);
    xo[1] = xi1 + sc * (xi1 - Oc[1] * inv);
    xo[2] = xi2 + sc * (xi2 - Oc[2] * inv);
  }
}

// ---------------------------------------------------------------------------
// h += ao @ Wo ; h += relu(h @ W1) @ W2     (one wave per row)
__global__ __launch_bounds__(32) void k_post(
    float* __restrict__ h, const float* __restrict__ ao,
    const float* __restrict__ Wo, const float* __restrict__ W1,
    const float* __restrict__ W2) {
  __shared__ float sA[Hq];
  __shared__ float sH[Hq];
  __shared__ float sM[2 * Hq];
  const long row = blockIdx.x;
  const int c = threadIdx.x;
  sA[c] = ao[row * Hq + c];
  __syncthreads();
  float s = 0.f;
#pragma unroll
  for (int e = 0; e < Hq; ++e) s += sA[e] * Wo[e * Hq + c];
  float hn = h[row * Hq + c] + s;
  sH[c] = hn;
  __syncthreads();
  float m0 = 0.f, m1 = 0.f;
#pragma unroll
  for (int e = 0; e < Hq; ++e) {
    float he = sH[e];
    m0 += he * W1[e * 2 * Hq + c];
    m1 += he * W1[e * 2 * Hq + c + Hq];
  }
  sM[c] = fmaxf(m0, 0.f);
  sM[c + Hq] = fmaxf(m1, 0.f);
  __syncthreads();
  float o = hn;
#pragma unroll
  for (int cc = 0; cc < 2 * Hq; ++cc) o += sM[cc] * W2[cc * Hq + c];
  h[row * Hq + c] = o;
}

// ---------------------------------------------------------------------------
// pooled mean over N, then tiny MLP head -> out [B,3]
__global__ __launch_bounds__(128) void k_pool(
    const float* __restrict__ h, const float* __restrict__ Wf1,
    const float* __restrict__ bf1, const float* __restrict__ Wf2,
    const float* __restrict__ bf2, float* __restrict__ out) {
  __shared__ float sp[Bq * Hq];
  __shared__ float sh[Bq * H2q];
  const int tid = threadIdx.x;
  const int b = tid >> 5, c = tid & 31;
  float s = 0.f;
  for (int n = 0; n < Nq; ++n) s += h[((long)b * Nq + n) * Hq + c];
  sp[b * Hq + c] = s * (1.0f / Nq);
  __syncthreads();
  if (tid < Bq * H2q) {
    int bb = tid >> 4, j = tid & 15;
    float a = bf1[j];
#pragma unroll
    for (int e = 0; e < Hq; ++e) a += sp[bb * Hq + e] * Wf1[e * H2q + j];
    sh[bb * H2q + j] = fmaxf(a, 0.f);
  }
  __syncthreads();
  if (tid < Bq * 3) {
    int bb = tid / 3, o = tid % 3;
    float a = bf2[o];
#pragma unroll
    for (int j = 0; j < H2q; ++j) a += sh[bb * H2q + j] * Wf2[j * 3 + o];
    out[bb * 3 + o] = a;
  }
}

// ---------------------------------------------------------------------------
extern "C" void kernel_launch(void* const* d_in, const int* in_sizes, int n_in,
                              void* d_out, int out_size, void* d_ws, size_t ws_size,
                              hipStream_t stream) {
  (void)in_sizes; (void)n_in; (void)out_size; (void)ws_size;
  const float* feats = (const float*)d_in[0];
  const float* coors = (const float*)d_in[1];
  const unsigned char* adj = (const unsigned char*)d_in[2];   // bool -> 1 byte
  const float* W_in  = (const float*)d_in[3];
  const float* Wq    = (const float*)d_in[4];
  const float* Wk    = (const float*)d_in[5];
  const float* Wv    = (const float*)d_in[6];
  const float* Wo    = (const float*)d_in[7];
  const float* W1    = (const float*)d_in[8];
  const float* W2    = (const float*)d_in[9];
  const float* cs    = (const float*)d_in[10];
  const float* Wf1   = (const float*)d_in[11];
  const float* bf1   = (const float*)d_in[12];
  const float* Wf2   = (const float*)d_in[13];
  const float* bf2   = (const float*)d_in[14];

  // workspace carve (all regions 256B aligned; ~4.4 MB total)
  char* p = (char*)d_ws;
  size_t off = 0;
  auto carve = [&](size_t bytes) { void* r = p + off; off = (off + bytes + 255) & ~(size_t)255; return r; };
  float*    h    = (float*)carve((size_t)BN * Hq * 4);
  float*    x0   = (float*)carve((size_t)BN * 3 * 4);
  float*    x1   = (float*)carve((size_t)BN * 3 * 4);
  _Float16* qf   = (_Float16*)carve((size_t)BN * Hq * 2);
  _Float16* kf   = (_Float16*)carve((size_t)BN * Hq * 2);
  _Float16* vT   = (_Float16*)carve((size_t)BN * Hq * 2);
  float*    ao   = (float*)carve((size_t)BN * Hq * 4);
  float*    g4   = (float*)carve((size_t)BN * 4 * 4);
  float*    gq4  = (float*)carve((size_t)BN * 4 * 4);
  _Float16* xT   = (_Float16*)carve((size_t)Bq * 16 * Nq * 2);

  k_embed<<<(BN * Hq) / 256, 256, 0, stream>>>(feats, coors, W_in, h, x0);

  float* xin = x0;
  float* xout = x1;
  for (int l = 0; l < Lq; ++l) {
    k_qkv<<<(BN * Hq) / 256, 256, 0, stream>>>(
        h, Wq + l * Hq * Hq, Wk + l * Hq * Hq, Wv + l * Hq * Hq, qf, kf, vT);
    k_geo<<<(Bq * 16 * Nq) / 256, 256, 0, stream>>>(xin, g4, gq4, xT);
    k_attn<<<BN / 16, 32, 0, stream>>>(qf, kf, vT, g4, gq4, xT, xin, adj,
                                       cs + l, ao, xout);
    k_post<<<BN, 32, 0, stream>>>(h, ao, Wo + l * Hq * Hq,
                                  W1 + l * Hq * 2 * Hq, W2 + l * 2 * Hq * Hq);
    float* t = xin; xin = xout; xout = t;
  }
  k_pool<<<1, 128, 0, stream>>>(h, Wf1, bf1, Wf2, bf2, (float*)d_out);
}
